// Attention_29721173688646
// MI455X (gfx1250) — matrix operand order repaired and checked by reading, not compiled, not run on hardware
//
#include <hip/hip_runtime.h>
#include <hip/hip_bf16.h>

// MI455X / gfx1250 attention forward: fp32 in -> bf16 WMMA GEMMs -> fp32 out.
// B=4, N=2048, C=1024, H=16, D=64. scale = H^-0.5 = 0.25 (folded into Q).

typedef __bf16 bf16_t;
typedef __attribute__((ext_vector_type(16))) __bf16 v16bf;
typedef __attribute__((ext_vector_type(8)))  float  v8f;
typedef __attribute__((ext_vector_type(4)))  unsigned tdm_g0_t;   // D# group 0 (4 SGPRs)
typedef __attribute__((ext_vector_type(8)))  unsigned tdm_g1_t;   // D# group 1 (8 SGPRs)

#define WMMA_BF16(A, B, C) \
  __builtin_amdgcn_wmma_f32_16x16x32_bf16(false, (A), false, (B), (short)0, (C), false, false)

// ---- CDNA5 async global->LDS copy (GLOBAL_LOAD_ASYNC_TO_LDS_B32, GV mode) ----
__device__ __forceinline__ void async_copy_b32(const float* g, float* lds) {
  unsigned lds_addr = (unsigned)(unsigned long long)lds;   // low 32 bits = LDS offset
  asm volatile("global_load_async_to_lds_b32 %0, %1, off"
               :: "v"(lds_addr), "v"((unsigned long long)g)
               : "memory");
}
__device__ __forceinline__ void wait_asynccnt0() {
  asm volatile("s_wait_asynccnt 0x0" ::: "memory");
}

// ---- CDNA5 Tensor Data Mover: DMA a [256 rows x 32 elems] bf16 tile from a
// K-major matrix (row stride 1024 elems) into LDS. 2-D descriptor per
// CDNA5 ISA (groups 0+1 only; VADDR2/3 disabled).
__device__ __forceinline__ void tdm_load_b_panel(const bf16_t* gsrc, unsigned lds_off,
                                                 unsigned rows_total) {
  const unsigned long long ga = (unsigned long long)gsrc;
  tdm_g0_t g0;
  g0[0] = 1u;                                                   // count=1 (valid user D#)
  g0[1] = lds_off;                                              // lds_addr (bytes)
  g0[2] = (unsigned)ga;                                         // global_addr[31:0]
  g0[3] = (unsigned)((ga >> 32) & 0x01FFFFFFull) | (2u << 30);  // ga[56:32] | type=2
  tdm_g1_t g1;
  g1[0] = 1u << 16;                              // wg_mask=0 | data_size=1 (2 bytes)
  g1[1] = 1024u << 16;                           // abar=0 | tensor_dim0[15:0]=1024
  g1[2] = (1024u >> 16) | (rows_total << 16);    // tensor_dim0[31:16] | tensor_dim1[15:0]
  g1[3] = (rows_total >> 16) | (32u << 16);      // tensor_dim1[31:16] | tile_dim0=32
  g1[4] = 256u;                                  // tile_dim1=256 | tile_dim2=0
  g1[5] = 1024u;                                 // tensor_dim0_stride[31:0]=1024
  g1[6] = 0u;                                    // stride0[47:32] | stride1[15:0]
  g1[7] = 0u;                                    // stride1[47:16]
  asm volatile("tensor_load_to_lds %0, %1" :: "s"(g0), "s"(g1) : "memory");
}

// ---- CDNA5 WMMA fragment loaders (wave32; element i holds k = 16*(lane/16) + i, same order for A and B) ----
__device__ __forceinline__ v16bf load_a_bf16(const bf16_t* __restrict__ A, int lda,
                                             int row0, int k0) {
  const int lane = threadIdx.x & 31;
  const bf16_t* p = A + (size_t)(row0 + (lane & 15)) * lda + k0 + ((lane & 16) ? 16 : 0);
  v16bf a;
#pragma unroll
  for (int i = 0; i < 16; ++i) a[i] = p[i];
  return a;
}

__device__ __forceinline__ v16bf load_b_kmajor(const bf16_t* __restrict__ Bt, int ldb,
                                               int col0, int k0) {
  const int lane = threadIdx.x & 31;
  const bf16_t* p = Bt + (size_t)(col0 + (lane & 15)) * ldb + k0 + ((lane & 16) ? 16 : 0);
  v16bf b;
#pragma unroll
  for (int i = 0; i < 16; ++i) b[i] = p[i];
  return b;
}

// B fragment from an LDS panel laid out [256 rows][32 k-elems] bf16.
__device__ __forceinline__ v16bf load_b_lds(const bf16_t* Bs, int colLocal) {
  const int lane = threadIdx.x & 31;
  const bf16_t* p = Bs + (size_t)(colLocal + (lane & 15)) * 32 + ((lane & 16) ? 16 : 0);
  v16bf b;
#pragma unroll
  for (int i = 0; i < 16; ++i) b[i] = p[i];
  return b;
}

// ---- conversion kernels ----
__global__ void cvt_bf16_kernel(const float* __restrict__ in, bf16_t* __restrict__ out, int n) {
  int i = (blockIdx.x * blockDim.x + threadIdx.x) * 4;
  if (i < n) {
    float4 v = *(const float4*)(in + i);
    out[i + 0] = (bf16_t)v.x;
    out[i + 1] = (bf16_t)v.y;
    out[i + 2] = (bf16_t)v.z;
    out[i + 3] = (bf16_t)v.w;
  }
}

// in[rows][cols] fp32 -> out[cols][rows] bf16 via async-staged 32x32 LDS tile.
__global__ __launch_bounds__(256) void transpose_cvt_kernel(const float* __restrict__ in,
                                                            bf16_t* __restrict__ out,
                                                            int rows, int cols) {
  __shared__ float T[32][33];
  const int tx = threadIdx.x & 31;
  const int ty = threadIdx.x >> 5;
  const int r0 = blockIdx.y * 32, c0 = blockIdx.x * 32;
#pragma unroll
  for (int k2 = 0; k2 < 4; ++k2) {
    int r = ty + k2 * 8;
    async_copy_b32(in + (size_t)(r0 + r) * cols + c0 + tx, &T[r][tx]);
  }
  wait_asynccnt0();
  __syncthreads();
#pragma unroll
  for (int k2 = 0; k2 < 4; ++k2) {
    int r = ty + k2 * 8;
    out[(size_t)(c0 + r) * rows + r0 + tx] = (bf16_t)T[tx][r];
  }
}

// ---- QKV GEMM with TDM-fed, double-buffered B panels in LDS ----
// 4 waves/WG; WG tile 32x256; wave tile 32x64 (8 WMMAs per K-step of 32).
__global__ __launch_bounds__(128) void qkv_gemm_kernel(const bf16_t* __restrict__ xb,
                                                       const bf16_t* __restrict__ wt,
                                                       bf16_t* __restrict__ qs,
                                                       bf16_t* __restrict__ kk,
                                                       bf16_t* __restrict__ vt) {
  __shared__ bf16_t Bs[2][256 * 32];                   // 2 x 16KB panels
  const int lane = threadIdx.x & 31;
  const int wave = threadIdx.x >> 5;
  const int row0 = blockIdx.x * 32;                    // M=8192
  const int colp = blockIdx.y * 256;                   // panel base over 3072
  const int col0 = colp + wave * 64;
  const bf16_t* wpanel = wt + (size_t)colp * 1024;
  v8f acc[2][4] = {{v8f{}, v8f{}, v8f{}, v8f{}}, {v8f{}, v8f{}, v8f{}, v8f{}}};

  if (wave == 0)
    tdm_load_b_panel(wpanel, (unsigned)(unsigned long long)&Bs[0][0], 3072u);
  for (int k0 = 0; k0 < 1024; k0 += 32) {
    const int cur = (k0 >> 5) & 1;
    if (wave == 0) __builtin_amdgcn_s_wait_tensorcnt(0);
    __syncthreads();                                   // panel `cur` visible to all
    if (wave == 0 && k0 + 32 < 1024)
      tdm_load_b_panel(wpanel + (k0 + 32),
                       (unsigned)(unsigned long long)&Bs[cur ^ 1][0], 3072u);
    __builtin_prefetch(xb + (size_t)(row0 + (lane & 15)) * 1024 + k0 + 64, 0, 3);
    v16bf a0 = load_a_bf16(xb, 1024, row0, k0);
    v16bf a1 = load_a_bf16(xb, 1024, row0 + 16, k0);
#pragma unroll
    for (int t = 0; t < 4; ++t) {
      v16bf b = load_b_lds(Bs[cur], wave * 64 + t * 16);
      acc[0][t] = WMMA_BF16(a0, b, acc[0][t]);
      acc[1][t] = WMMA_BF16(a1, b, acc[1][t]);
    }
    __syncthreads();                                   // done with `cur` before refill
  }
  const int c = lane & 15;
  const int rbase = (lane & 16) ? 8 : 0;
#pragma unroll
  for (int rs = 0; rs < 2; ++rs) {
#pragma unroll
    for (int t = 0; t < 4; ++t) {
      int col = col0 + t * 16 + c;
      int three = col >> 10;                           // 0=q 1=k 2=v
      int h = (col & 1023) >> 6;
      int d = col & 63;
#pragma unroll
      for (int j = 0; j < 8; ++j) {
        int m = row0 + rs * 16 + rbase + j;
        int bi = m >> 11, n = m & 2047;
        float v = acc[rs][t][j];
        size_t bhnd = (((size_t)bi * 16 + h) * 2048 + n) * 64 + d;
        if (three == 0)      qs[bhnd] = (bf16_t)(v * 0.25f);  // scale folded into Q
        else if (three == 1) kk[bhnd] = (bf16_t)v;
        else                 vt[(((size_t)bi * 16 + h) * 64 + d) * 2048 + n] = (bf16_t)v;
      }
    }
  }
}

// ---- attention: 4 waves/WG cooperate on one (b,h,16-query block) ----
__global__ __launch_bounds__(128) void attn_kernel(const bf16_t* __restrict__ qs,
                                                   const bf16_t* __restrict__ kk,
                                                   const bf16_t* __restrict__ vt,
                                                   bf16_t* __restrict__ out) {
  extern __shared__ float smem[];
  float* S      = smem;                 // 16*2048 f32 (scores, then unnormalized P)
  float* redmax = S + 16 * 2048;        // [8 segs][16 rows]
  float* redsum = redmax + 128;
  float* rowmax = redsum + 128;         // [16]
  float* rowinv = rowmax + 16;          // [16]
  float* Opart  = rowinv + 16;          // [4 waves][16][64]
  const int tid  = threadIdx.x;
  const int lane = tid & 31;
  const int wave = tid >> 5;
  const int qblk = blockIdx.x & 127;
  const int bh   = blockIdx.x >> 7;
  const bf16_t* q  = qs + (size_t)bh * 2048 * 64;
  const bf16_t* kp = kk + (size_t)bh * 2048 * 64;
  const bf16_t* vp = vt + (size_t)bh * 64 * 2048;
  const int c = lane & 15;
  const int rbase = (lane & 16) ? 8 : 0;

  // S = (Q*scale) @ K^T : wave w handles key tiles [w*32, w*32+32)
  v16bf aq0 = load_a_bf16(q, 64, qblk * 16, 0);
  v16bf aq1 = load_a_bf16(q, 64, qblk * 16, 32);
  for (int kt = wave * 32; kt < wave * 32 + 32; ++kt) {
    __builtin_prefetch(kp + (size_t)((kt + 4) * 16 + (lane & 15)) * 64, 0, 3);
    v8f s = v8f{};
    v16bf b0 = load_b_kmajor(kp, 64, kt * 16, 0);
    s = WMMA_BF16(aq0, b0, s);
    v16bf b1 = load_b_kmajor(kp, 64, kt * 16, 32);
    s = WMMA_BF16(aq1, b1, s);
#pragma unroll
    for (int j = 0; j < 8; ++j) S[(rbase + j) * 2048 + kt * 16 + c] = s[j];
  }
  __syncthreads();

  // softmax: tid -> (row = tid&15, seg = tid>>4); each scans 256 cols
  const int srow = tid & 15, seg = tid >> 4;
  float* sp = S + srow * 2048 + seg * 256;
  float mx = -3.402823466e38f;
  for (int i = 0; i < 256; ++i) mx = fmaxf(mx, sp[i]);
  redmax[seg * 16 + srow] = mx;
  __syncthreads();
  if (tid < 16) {
    float m = redmax[tid];
#pragma unroll
    for (int s2 = 1; s2 < 8; ++s2) m = fmaxf(m, redmax[s2 * 16 + tid]);
    rowmax[tid] = m;
  }
  __syncthreads();
  const float rm = rowmax[srow];
  float sum = 0.f;
  for (int i = 0; i < 256; ++i) {
    float e = __expf(sp[i] - rm);
    sp[i] = e;
    sum += e;
  }
  redsum[seg * 16 + srow] = sum;
  __syncthreads();
  if (tid < 16) {
    float s2 = redsum[tid];
#pragma unroll
    for (int k2 = 1; k2 < 8; ++k2) s2 += redsum[k2 * 16 + tid];
    rowinv[tid] = 1.0f / s2;
  }
  __syncthreads();

  // O_partial = P @ V : wave w handles keys [w*512, w*512+512)
  v8f o[4] = {v8f{}, v8f{}, v8f{}, v8f{}};
  const int row = lane & 15;
  for (int kc = 0; kc < 16; ++kc) {
    const int kbase = wave * 512 + kc * 32;
    const float* prow = S + row * 2048 + kbase + ((lane & 16) ? 16 : 0);
    v16bf a;
#pragma unroll
    for (int i = 0; i < 16; ++i) a[i] = (bf16_t)prow[i];
#pragma unroll
    for (int t = 0; t < 4; ++t) {
      v16bf b = load_b_kmajor(vp, 2048, t * 16, kbase);
      o[t] = WMMA_BF16(a, b, o[t]);
    }
  }
#pragma unroll
  for (int t = 0; t < 4; ++t)
#pragma unroll
    for (int j = 0; j < 8; ++j)
      Opart[wave * 1024 + (rbase + j) * 64 + t * 16 + c] = o[t][j];
  __syncthreads();

  // combine partials, normalize, store [B,N,H,D] bf16
  const int bI = bh >> 4, h = bh & 15;
#pragma unroll
  for (int u = 0; u < 8; ++u) {
    int e = tid * 8 + u;
    int r = e >> 6, d = e & 63;
    float v = (Opart[e] + Opart[1024 + e] + Opart[2048 + e] + Opart[3072 + e]) * rowinv[r];
    int n = qblk * 16 + r;
    out[(((size_t)bI * 2048 + n) * 16 + h) * 64 + d] = (bf16_t)v;
  }
}

// ---- output projection with TDM-fed B panels: out = attn @ Wproj_t + b ----
__global__ __launch_bounds__(128) void proj_gemm_kernel(const bf16_t* __restrict__ xa,
                                                        const bf16_t* __restrict__ wt,
                                                        const float* __restrict__ bias,
                                                        float* __restrict__ out) {
  __shared__ bf16_t Bs[2][256 * 32];
  const int lane = threadIdx.x & 31;
  const int wave = threadIdx.x >> 5;
  const int row0 = blockIdx.x * 32;
  const int colp = blockIdx.y * 256;
  const int col0 = colp + wave * 64;
  const bf16_t* wpanel = wt + (size_t)colp * 1024;
  v8f acc[2][4] = {{v8f{}, v8f{}, v8f{}, v8f{}}, {v8f{}, v8f{}, v8f{}, v8f{}}};

  if (wave == 0)
    tdm_load_b_panel(wpanel, (unsigned)(unsigned long long)&Bs[0][0], 1024u);
  for (int k0 = 0; k0 < 1024; k0 += 32) {
    const int cur = (k0 >> 5) & 1;
    if (wave == 0) __builtin_amdgcn_s_wait_tensorcnt(0);
    __syncthreads();
    if (wave == 0 && k0 + 32 < 1024)
      tdm_load_b_panel(wpanel + (k0 + 32),
                       (unsigned)(unsigned long long)&Bs[cur ^ 1][0], 1024u);
    __builtin_prefetch(xa + (size_t)(row0 + (lane & 15)) * 1024 + k0 + 64, 0, 3);
    v16bf a0 = load_a_bf16(xa, 1024, row0, k0);
    v16bf a1 = load_a_bf16(xa, 1024, row0 + 16, k0);
#pragma unroll
    for (int t = 0; t < 4; ++t) {
      v16bf b = load_b_lds(Bs[cur], wave * 64 + t * 16);
      acc[0][t] = WMMA_BF16(a0, b, acc[0][t]);
      acc[1][t] = WMMA_BF16(a1, b, acc[1][t]);
    }
    __syncthreads();
  }
  const int c = lane & 15;
  const int rbase = (lane & 16) ? 8 : 0;
#pragma unroll
  for (int rs = 0; rs < 2; ++rs) {
#pragma unroll
    for (int t = 0; t < 4; ++t) {
      int col = col0 + t * 16 + c;
      float bv = bias[col];
#pragma unroll
      for (int j = 0; j < 8; ++j) {
        int m = row0 + rs * 16 + rbase + j;
        out[(size_t)m * 1024 + col] = acc[rs][t][j] + bv;
      }
    }
  }
}

extern "C" void kernel_launch(void* const* d_in, const int* in_sizes, int n_in,
                              void* d_out, int out_size, void* d_ws, size_t ws_size,
                              hipStream_t stream) {
  const float* x     = (const float*)d_in[0];   // [4,2048,1024]
  const float* Wqkv  = (const float*)d_in[1];   // [1024,3072]
  const float* Wproj = (const float*)d_in[2];   // [1024,1024]
  const float* bproj = (const float*)d_in[3];   // [1024]
  float* out = (float*)d_out;

  char* ws = (char*)d_ws;
  size_t off = 0;
  auto wsAlloc = [&](size_t bytes) -> void* {
    void* p = ws + off;
    off = (off + bytes + 255) & ~(size_t)255;
    return p;
  };
  const int XN = 4 * 2048 * 1024;               // 8388608
  bf16_t* xb     = (bf16_t*)wsAlloc((size_t)XN * 2);
  bf16_t* wqkvt  = (bf16_t*)wsAlloc((size_t)1024 * 3072 * 2);
  bf16_t* wprojt = (bf16_t*)wsAlloc((size_t)1024 * 1024 * 2);
  bf16_t* q      = (bf16_t*)wsAlloc((size_t)XN * 2);  // [B,H,N,D], pre-scaled
  bf16_t* k      = (bf16_t*)wsAlloc((size_t)XN * 2);  // [B,H,N,D]
  bf16_t* vt     = (bf16_t*)wsAlloc((size_t)XN * 2);  // [B,H,D,N]
  bf16_t* att    = (bf16_t*)wsAlloc((size_t)XN * 2);  // [B,N,H,D]

  cvt_bf16_kernel<<<XN / 4 / 256, 256, 0, stream>>>(x, xb, XN);
  transpose_cvt_kernel<<<dim3(3072 / 32, 1024 / 32), 256, 0, stream>>>(Wqkv, wqkvt, 1024, 3072);
  transpose_cvt_kernel<<<dim3(1024 / 32, 1024 / 32), 256, 0, stream>>>(Wproj, wprojt, 1024, 1024);

  qkv_gemm_kernel<<<dim3(256, 12), 128, 0, stream>>>(xb, wqkvt, q, k, vt);

  const size_t attn_lds = (size_t)(16 * 2048 + 128 + 128 + 16 + 16 + 4 * 16 * 64) * 4;
  attn_kernel<<<4 * 16 * 128, 128, attn_lds, stream>>>(q, k, vt, att);

  proj_gemm_kernel<<<dim3(256, 4), 128, 0, stream>>>(att, wprojt, bproj, out);
}
